// GPT_65592740545146
// MI455X (gfx1250) — matrix operand order repaired and checked by reading, not compiled, not run on hardware
//
#include <hip/hip_runtime.h>
#include <hip/hip_bf16.h>
#include <stdint.h>

// ---------------------------------------------------------------------------
// modded-nanogpt forward pass for MI455X (gfx1250, wave32, WMMA).
// All GEMMs use v_wmma_f32_16x16x32_bf16 (f32 -> bf16 RNE at tile load,
// f32 accumulation). Attention is a flash-style online-softmax kernel with
// WMMA for both Q.K^T and P.V, mask = causal & same-doc & block window.
// ---------------------------------------------------------------------------

#define TSEQ   2048
#define VOCAB  50257
#define VPAD   50304
#define DMODEL 768
#define NHEAD  6
#define HDIM   128
#define NLAYER 12
#define EPSRMS 1.1920929e-07f
#define LDSD   34   // padded LDS row stride in bf16 elems (32 data + 2 pad)

typedef __bf16 bf16_t;
typedef bf16_t bf16x16 __attribute__((ext_vector_type(16)));
typedef float  f32x8   __attribute__((ext_vector_type(8)));

union FragAB { unsigned u[8]; bf16x16 v; };
union FragC  { float    f[8]; f32x8   v; };

__device__ __forceinline__ unsigned short f2bf(float f) {
  unsigned a = __builtin_bit_cast(unsigned, f);
  a += 0x7FFFu + ((a >> 16) & 1u);              // round-to-nearest-even
  return (unsigned short)(a >> 16);
}
__device__ __forceinline__ unsigned pack2bf(float lo, float hi) {
  return (unsigned)f2bf(lo) | ((unsigned)f2bf(hi) << 16);
}

// ---------------------------------------------------------------------------
// Generic GEMM: C[M,N] = A[M,K] * B[N,K]^T   (A,B,C f32 row-major, K%32==0,
// M,N multiples of 128).  8 waves/block, each wave owns a 32x64 sub-tile.
// ---------------------------------------------------------------------------
__global__ __launch_bounds__(256) void gemm_xwT(const float* __restrict__ A,
                                                const float* __restrict__ B,
                                                float* __restrict__ C,
                                                int M, int N, int K) {
  __shared__ unsigned short ldsA[128 * LDSD];
  __shared__ unsigned short ldsB[128 * LDSD];
  const int tid  = threadIdx.x;
  const int lane = tid & 31, wave = tid >> 5;
  const int hs   = lane >> 4, lm = lane & 15;
  const int bm = blockIdx.y * 128, bn = blockIdx.x * 128;
  const int wmr = (wave & 3) * 32;   // wave row offset within tile
  const int wnc = (wave >> 2) * 64;  // wave col offset within tile
  (void)M;

  FragC acc[2][4];
#pragma unroll
  for (int i = 0; i < 2; ++i)
#pragma unroll
    for (int j = 0; j < 4; ++j)
#pragma unroll
      for (int e = 0; e < 8; ++e) acc[i][j].f[e] = 0.0f;

  for (int k0 = 0; k0 < K; k0 += 32) {
    // Cooperative tile load: 128x32 f32 -> packed bf16 pairs in LDS.
#pragma unroll
    for (int j = 0; j < 8; ++j) {
      int pidx = tid + 256 * j;
      int row  = pidx >> 4;
      int kp   = (pidx & 15) * 2;
      float2 av = *(const float2*)(A + (size_t)(bm + row) * K + k0 + kp);
      *(unsigned*)(ldsA + row * LDSD + kp) = pack2bf(av.x, av.y);
      float2 bv = *(const float2*)(B + (size_t)(bn + row) * K + k0 + kp);
      *(unsigned*)(ldsB + row * LDSD + kp) = pack2bf(bv.x, bv.y);
    }
    __syncthreads();

    FragAB afr[2], bfr[4];
#pragma unroll
    for (int mt = 0; mt < 2; ++mt) {
      int r = wmr + mt * 16 + lm;
#pragma unroll
      for (int v = 0; v < 8; ++v) {
        int kk = (v < 4 ? 2 * v : 16 + 2 * (v - 4)) + 8 * hs;  // A-frag K map
        afr[mt].u[v] = *(const unsigned*)(ldsA + r * LDSD + kk);
      }
    }
#pragma unroll
    for (int nt = 0; nt < 4; ++nt) {
      int c = wnc + nt * 16 + lm;
#pragma unroll
      for (int v = 0; v < 8; ++v)
        bfr[nt].u[v] = *(const unsigned*)(ldsB + c * LDSD + (v < 4 ? 2 * v : 16 + 2 * (v - 4)) + 8 * hs);
    }
#pragma unroll
    for (int mt = 0; mt < 2; ++mt)
#pragma unroll
      for (int nt = 0; nt < 4; ++nt)
        acc[mt][nt].v = __builtin_amdgcn_wmma_f32_16x16x32_bf16(
            false, afr[mt].v, false, bfr[nt].v, (short)0, acc[mt][nt].v,
            false, false);
    __syncthreads();
  }

#pragma unroll
  for (int mt = 0; mt < 2; ++mt)
#pragma unroll
    for (int nt = 0; nt < 4; ++nt)
#pragma unroll
      for (int v = 0; v < 8; ++v)
        C[(size_t)(bm + wmr + mt * 16 + hs * 8 + v) * N + bn + wnc + nt * 16 + lm] =
            acc[mt][nt].f[v];
}

// ---------------------------------------------------------------------------
// Flash attention: one wave per (head, 16-query tile). Online softmax over
// 32-key tiles, mask = causal & same-doc & (qblk-kblk < W).
// ---------------------------------------------------------------------------
__global__ __launch_bounds__(256) void flash_attn(const float* __restrict__ Qm,
                                                  const float* __restrict__ Km,
                                                  const float* __restrict__ Vm,
                                                  float* __restrict__ Om,
                                                  const int* __restrict__ docs,
                                                  const int* __restrict__ winp) {
  __shared__ unsigned short plds[8 * 16 * LDSD];  // per-wave P staging
  const int tid = threadIdx.x, lane = tid & 31, wv = tid >> 5;
  const int hs = lane >> 4, lm = lane & 15;
  const int gw = blockIdx.x * 8 + wv;
  const int h  = gw / (TSEQ / 16);
  const int q0 = (gw % (TSEQ / 16)) * 16;
  const int W  = winp[0];
  unsigned short* pw = plds + wv * 16 * LDSD;
  const float scale = 0.08838834764831845f;  // 128^-0.5 folded into Q

  FragAB qf[4];
#pragma unroll
  for (int c = 0; c < 4; ++c)
#pragma unroll
    for (int v = 0; v < 8; ++v) {
      int kk = (v < 4 ? 2 * v : 16 + 2 * (v - 4)) + 8 * hs + 32 * c;
      float2 t2 = *(const float2*)(Qm + (size_t)(q0 + lm) * DMODEL + h * HDIM + kk);
      qf[c].u[v] = pack2bf(t2.x * scale, t2.y * scale);
    }
  int docq[8];
#pragma unroll
  for (int v = 0; v < 8; ++v) docq[v] = docs[q0 + hs * 8 + v];

  float mrow[8], lrow[8];
  FragC acc[8];
#pragma unroll
  for (int v = 0; v < 8; ++v) { mrow[v] = -3.0e38f; lrow[v] = 0.0f; }
#pragma unroll
  for (int dt = 0; dt < 8; ++dt)
#pragma unroll
    for (int e = 0; e < 8; ++e) acc[dt].f[e] = 0.0f;

  int j0min = (q0 >> 7) - W + 1;
  j0min = (j0min < 0 ? 0 : j0min) << 7;

  for (int j0 = j0min; j0 < q0 + 16; j0 += 32) {
    // ---- S = Q.K^T over 128 dims, two 16x16 key tiles ----
    FragC s[2];
#pragma unroll
    for (int nt = 0; nt < 2; ++nt)
#pragma unroll
      for (int e = 0; e < 8; ++e) s[nt].f[e] = 0.0f;
#pragma unroll
    for (int nt = 0; nt < 2; ++nt)
#pragma unroll
      for (int c = 0; c < 4; ++c) {
        FragAB kf;
#pragma unroll
        for (int v = 0; v < 8; ++v) {
          int kk = 32 * c + (v < 4 ? 2 * v : 16 + 2 * (v - 4)) + 8 * hs;
          float2 t2 = *(const float2*)(Km + (size_t)(j0 + nt * 16 + lm) * DMODEL +
                                       h * HDIM + kk);
          kf.u[v] = pack2bf(t2.x, t2.y);
        }
        s[nt].v = __builtin_amdgcn_wmma_f32_16x16x32_bf16(
            false, qf[c].v, false, kf.v, (short)0, s[nt].v, false, false);
      }

    // ---- mask + online softmax ----
#pragma unroll
    for (int nt = 0; nt < 2; ++nt) {
      int key  = j0 + nt * 16 + lm;
      int dock = docs[key];
#pragma unroll
      for (int v = 0; v < 8; ++v) {
        int qr = q0 + hs * 8 + v;
        bool ok = (key <= qr) && (dock == docq[v]) &&
                  (((qr >> 7) - (key >> 7)) < W);
        s[nt].f[v] = ok ? s[nt].f[v] : -3.0e38f;
      }
    }
    float alpha[8];
#pragma unroll
    for (int v = 0; v < 8; ++v) {
      float mx = fmaxf(s[0].f[v], s[1].f[v]);
#pragma unroll
      for (int off = 1; off < 16; off <<= 1)
        mx = fmaxf(mx, __shfl_xor(mx, off, 32));
      float mnew = fmaxf(mrow[v], mx);
      float p0 = (s[0].f[v] < -1.0e38f) ? 0.0f : __expf(s[0].f[v] - mnew);
      float p1 = (s[1].f[v] < -1.0e38f) ? 0.0f : __expf(s[1].f[v] - mnew);
      s[0].f[v] = p0;
      s[1].f[v] = p1;
      float ps = p0 + p1;
#pragma unroll
      for (int off = 1; off < 16; off <<= 1) ps += __shfl_xor(ps, off, 32);
      alpha[v] = (mrow[v] < -1.0e38f) ? 0.0f : __expf(mrow[v] - mnew);
      lrow[v]  = lrow[v] * alpha[v] + ps;
      mrow[v]  = mnew;
    }

    // ---- P (C-layout) -> LDS -> A fragment ----
#pragma unroll
    for (int nt = 0; nt < 2; ++nt)
#pragma unroll
      for (int v = 0; v < 8; ++v)
        pw[(hs * 8 + v) * LDSD + nt * 16 + lm] = f2bf(s[nt].f[v]);
    asm volatile("s_wait_dscnt 0" ::: "memory");
    FragAB pf;
#pragma unroll
    for (int v = 0; v < 8; ++v) {
      int kk = (v < 4 ? 2 * v : 16 + 2 * (v - 4)) + 8 * hs;
      pf.u[v] = *(const unsigned*)(pw + lm * LDSD + kk);
    }

    // ---- O = O*alpha + P.V ----
#pragma unroll
    for (int dt = 0; dt < 8; ++dt) {
#pragma unroll
      for (int v = 0; v < 8; ++v) acc[dt].f[v] *= alpha[v];
      FragAB vf;
#pragma unroll
      for (int j = 0; j < 8; ++j) {
        int key0 = j0 + (j < 4 ? 2 * j : 16 + 2 * (j - 4)) + 8 * hs;
        float x0 = Vm[(size_t)key0 * DMODEL + h * HDIM + dt * 16 + lm];
        float x1 = Vm[(size_t)(key0 + 1) * DMODEL + h * HDIM + dt * 16 + lm];
        vf.u[j] = pack2bf(x0, x1);
      }
      acc[dt].v = __builtin_amdgcn_wmma_f32_16x16x32_bf16(
          false, pf.v, false, vf.v, (short)0, acc[dt].v, false, false);
    }
  }

#pragma unroll
  for (int dt = 0; dt < 8; ++dt)
#pragma unroll
    for (int v = 0; v < 8; ++v)
      Om[(size_t)(q0 + hs * 8 + v) * DMODEL + h * HDIM + dt * 16 + lm] =
          acc[dt].f[v] / lrow[v];
}

// ---------------------------------------------------------------------------
// Per-head RMSNorm + rotary for q and k.  One wave per (t, head).
// ---------------------------------------------------------------------------
__global__ __launch_bounds__(256) void qk_rope(float* __restrict__ Qm,
                                               float* __restrict__ Km) {
  const int tid = threadIdx.x, lane = tid & 31, wv = tid >> 5;
  const int gw = blockIdx.x * 8 + wv;
  const int t = gw / NHEAD, h = gw % NHEAD;
#pragma unroll
  for (int which = 0; which < 2; ++which) {
    float* base = (which ? Km : Qm) + (size_t)t * DMODEL + h * HDIM;
    float x[4];
#pragma unroll
    for (int c = 0; c < 4; ++c) x[c] = base[lane + 32 * c];
    float ss = x[0] * x[0] + x[1] * x[1] + x[2] * x[2] + x[3] * x[3];
#pragma unroll
    for (int off = 1; off < 32; off <<= 1) ss += __shfl_xor(ss, off, 32);
    float r = rsqrtf(ss * (1.0f / HDIM) + EPSRMS);
#pragma unroll
    for (int c = 0; c < 4; ++c) x[c] *= r;
    float out[4];
#pragma unroll
    for (int c = 0; c < 2; ++c) {
      int d1 = lane + 32 * c;
      // freq = (1/1024)^(d1/31) for d1<32 else 0
      float fr = (d1 < 32) ? __expf(-0.22359586469952337f * (float)d1) : 0.0f;
      float th = (float)t * fr;
      float si, co;
      __sincosf(th, &si, &co);
      out[c]     = x[c] * co + x[c + 2] * si;
      out[c + 2] = -x[c] * si + x[c + 2] * co;
    }
#pragma unroll
    for (int c = 0; c < 4; ++c) base[lane + 32 * c] = out[c];
  }
}

// ---------------------------------------------------------------------------
// Row RMSNorm over D, one block per token.
// ---------------------------------------------------------------------------
__global__ __launch_bounds__(256) void rmsnorm_k(const float* __restrict__ X,
                                                 float* __restrict__ Y, int Dd) {
  __shared__ float red[256];
  const int t = blockIdx.x, tid = threadIdx.x;
  const float* x = X + (size_t)t * Dd;
  float s = 0.0f;
  for (int i = tid; i < Dd; i += 256) s += x[i] * x[i];
  red[tid] = s; __syncthreads();
  for (int o = 128; o > 0; o >>= 1) {
    if (tid < o) red[tid] += red[tid + o];
    __syncthreads();
  }
  float r = rsqrtf(red[0] / (float)Dd + EPSRMS);
  float* y = Y + (size_t)t * Dd;
  for (int i = tid; i < Dd; i += 256) y[i] = x[i] * r;
}

__global__ __launch_bounds__(256) void embed_norm(const float* __restrict__ emb,
                                                  const int* __restrict__ inp,
                                                  float* __restrict__ X,
                                                  float* __restrict__ X0) {
  __shared__ float red[256];
  const int t = blockIdx.x, tid = threadIdx.x;
  const float* x = emb + (size_t)inp[t] * DMODEL;
  float s = 0.0f;
  for (int i = tid; i < DMODEL; i += 256) s += x[i] * x[i];
  red[tid] = s; __syncthreads();
  for (int o = 128; o > 0; o >>= 1) {
    if (tid < o) red[tid] += red[tid + o];
    __syncthreads();
  }
  float r = rsqrtf(red[0] / (float)DMODEL + EPSRMS);
  for (int i = tid; i < DMODEL; i += 256) {
    float v = x[i] * r;
    X[(size_t)t * DMODEL + i]  = v;
    X0[(size_t)t * DMODEL + i] = v;
  }
}

// ---------------------------------------------------------------------------
// Elementwise helpers.
// ---------------------------------------------------------------------------
__global__ void mix_k(float* __restrict__ X, const float* __restrict__ X0,
                      const float* __restrict__ lam, int n) {
  int i = blockIdx.x * 256 + threadIdx.x;
  if (i < n) X[i] = lam[0] * X[i] + lam[1] * X0[i];
}
__global__ void vmix_k(float* __restrict__ Vm, const float* __restrict__ vemb,
                       const int* __restrict__ inp, const float* __restrict__ lam) {
  int i = blockIdx.x * 256 + threadIdx.x;
  if (i < TSEQ * DMODEL) {
    int t = i / DMODEL, d = i - t * DMODEL;
    float ve = vemb ? vemb[(size_t)inp[t] * DMODEL + d] : 0.0f;
    Vm[i] = lam[0] * Vm[i] + lam[1] * ve;
  }
}
__global__ void add_k(float* __restrict__ X, const float* __restrict__ Ym, int n) {
  int i = blockIdx.x * 256 + threadIdx.x;
  if (i < n) X[i] += Ym[i];
}
__global__ void addscale_k(float* __restrict__ X, const float* __restrict__ Ym,
                           const float* __restrict__ sw, int si, int n) {
  int i = blockIdx.x * 256 + threadIdx.x;
  if (i < n) X[i] += sw[si] * Ym[i];
}
__global__ void relu2_k(float* __restrict__ H, int n) {
  int i = blockIdx.x * 256 + threadIdx.x;
  if (i < n) { float v = fmaxf(H[i], 0.0f); H[i] = v * v; }
}
__global__ void docs_k(const int* __restrict__ inp, int* __restrict__ docs) {
  if (threadIdx.x == 0 && blockIdx.x == 0) {
    int c = 0;
    for (int t = 0; t < TSEQ; ++t) { c += (inp[t] == 50256); docs[t] = c; }
  }
}

// ---------------------------------------------------------------------------
// Loss: tanh-capped logits -> log-softmax over VPAD -> -mean logp[target].
// ---------------------------------------------------------------------------
__global__ __launch_bounds__(256) void loss_k(const float* __restrict__ logits,
                                              const int* __restrict__ tgt,
                                              float* __restrict__ out) {
  __shared__ float red[256];
  const int t = blockIdx.x, tid = threadIdx.x;
  const float* row = logits + (size_t)t * VPAD;
  float mx = -3.0e38f;
  for (int i = tid; i < VPAD; i += 256) {
    float z = 30.0f * tanhf(row[i] * (1.0f / 30.0f));
    mx = fmaxf(mx, z);
  }
  red[tid] = mx; __syncthreads();
  for (int o = 128; o > 0; o >>= 1) {
    if (tid < o) red[tid] = fmaxf(red[tid], red[tid + o]);
    __syncthreads();
  }
  float rm = red[0];
  __syncthreads();
  float se = 0.0f;
  for (int i = tid; i < VPAD; i += 256) {
    float z = 30.0f * tanhf(row[i] * (1.0f / 30.0f));
    se += __expf(z - rm);
  }
  red[tid] = se; __syncthreads();
  for (int o = 128; o > 0; o >>= 1) {
    if (tid < o) red[tid] += red[tid + o];
    __syncthreads();
  }
  if (tid == 0) {
    float zt = 30.0f * tanhf(row[tgt[t]] * (1.0f / 30.0f));
    float lp = zt - rm - __logf(red[0]);
    atomicAdd(out, -lp * (1.0f / TSEQ));
  }
}

// ---------------------------------------------------------------------------
// Host orchestration.
// ---------------------------------------------------------------------------
extern "C" void kernel_launch(void* const* d_in, const int* in_sizes, int n_in,
                              void* d_out, int out_size, void* d_ws, size_t ws_size,
                              hipStream_t stream) {
  struct Blk {
    const float *lambdas, *w_fc, *w_fcproj, *wq, *wk, *wv, *wo, *attn_lambdas;
    bool attn;
  };
  const float *embed = nullptr, *vembeds = nullptr, *skipw = nullptr, *lmhead = nullptr;
  const int *inputs = nullptr, *targets = nullptr, *winp = nullptr;
  Blk blk[NLAYER] = {};
  int p = 0;
  // jax tree-flatten sorts dict keys -> first leaf is 'inputs' (2048).
  // Plain dict-insertion order -> first leaf is params.embed (38,597,376).
  bool sortedOrder = (in_sizes[0] == TSEQ);
  if (sortedOrder) {
    inputs = (const int*)d_in[p++];
    for (int i = 0; i < NLAYER; ++i) {
      if (i != 7) {
        blk[i].attn = true;
        blk[i].attn_lambdas = (const float*)d_in[p++];
        blk[i].lambdas      = (const float*)d_in[p++];
        blk[i].w_fc         = (const float*)d_in[p++];
        blk[i].w_fcproj     = (const float*)d_in[p++];
        blk[i].wk           = (const float*)d_in[p++];
        blk[i].wo           = (const float*)d_in[p++];
        blk[i].wq           = (const float*)d_in[p++];
        blk[i].wv           = (const float*)d_in[p++];
      } else {
        blk[i].lambdas  = (const float*)d_in[p++];
        blk[i].w_fc     = (const float*)d_in[p++];
        blk[i].w_fcproj = (const float*)d_in[p++];
      }
    }
    embed   = (const float*)d_in[p++];
    lmhead  = (const float*)d_in[p++];
    skipw   = (const float*)d_in[p++];
    vembeds = (const float*)d_in[p++];
    winp    = (const int*)d_in[p++];
    targets = (const int*)d_in[p++];
  } else {
    embed   = (const float*)d_in[p++];
    vembeds = (const float*)d_in[p++];
    for (int i = 0; i < NLAYER; ++i) {
      blk[i].lambdas  = (const float*)d_in[p++];
      blk[i].w_fc     = (const float*)d_in[p++];
      blk[i].w_fcproj = (const float*)d_in[p++];
      if (i != 7) {
        blk[i].attn = true;
        blk[i].wq           = (const float*)d_in[p++];
        blk[i].wk           = (const float*)d_in[p++];
        blk[i].wv           = (const float*)d_in[p++];
        blk[i].wo           = (const float*)d_in[p++];
        blk[i].attn_lambdas = (const float*)d_in[p++];
      }
    }
    skipw   = (const float*)d_in[p++];
    lmhead  = (const float*)d_in[p++];
    inputs  = (const int*)d_in[p++];
    targets = (const int*)d_in[p++];
    winp    = (const int*)d_in[p++];
  }
  (void)n_in;

  // Workspace bump allocator.
  char* ws = (char*)d_ws;
  size_t off = 0;
  auto alloc = [&](size_t bytes) -> void* {
    void* r = (void*)(ws + off);
    off += (bytes + 255) & ~(size_t)255;
    return r;
  };
  const size_t TD = (size_t)TSEQ * DMODEL;
  float* x   = (float*)alloc(TD * 4);
  float* x0  = (float*)alloc(TD * 4);
  float* xn  = (float*)alloc(TD * 4);
  float* qb  = (float*)alloc(TD * 4);
  float* kb  = (float*)alloc(TD * 4);
  float* vb  = (float*)alloc(TD * 4);
  float* yb  = (float*)alloc(TD * 4);
  float* tmp = (float*)alloc(TD * 4);
  float* hb  = (float*)alloc(TD * 4 * 4);
  float* skips[6];
  for (int i = 0; i < 6; ++i) skips[i] = (float*)alloc(TD * 4);
  int*   docs   = (int*)alloc(TSEQ * 4);
  float* logits = (float*)alloc((size_t)TSEQ * VPAD * 4);
  (void)ws_size;

  hipMemsetAsync(d_out, 0, sizeof(float) * (size_t)out_size, stream);

  docs_k<<<1, 32, 0, stream>>>(inputs, docs);
  embed_norm<<<TSEQ, 256, 0, stream>>>(embed, inputs, x, x0);

  const int EW = (int)(TD / 256);
  dim3 g768(DMODEL / 128, TSEQ / 128);
  dim3 g3072(4 * DMODEL / 128, TSEQ / 128);
  dim3 gvp(VPAD / 128, TSEQ / 128);

  for (int i = 0; i < NLAYER; ++i) {
    if (i >= 6)
      addscale_k<<<EW, 256, 0, stream>>>(x, skips[11 - i], skipw, i - 6, (int)TD);
    mix_k<<<EW, 256, 0, stream>>>(x, x0, blk[i].lambdas, (int)TD);
    if (blk[i].attn) {
      rmsnorm_k<<<TSEQ, 256, 0, stream>>>(x, xn, DMODEL);
      gemm_xwT<<<g768, 256, 0, stream>>>(xn, blk[i].wq, qb, TSEQ, DMODEL, DMODEL);
      gemm_xwT<<<g768, 256, 0, stream>>>(xn, blk[i].wk, kb, TSEQ, DMODEL, DMODEL);
      gemm_xwT<<<g768, 256, 0, stream>>>(xn, blk[i].wv, vb, TSEQ, DMODEL, DMODEL);
      qk_rope<<<TSEQ * NHEAD / 8, 256, 0, stream>>>(qb, kb);
      const float* ve = nullptr;
      if (i < 3)       ve = vembeds + (size_t)i * VOCAB * DMODEL;
      else if (i >= 9) ve = vembeds + (size_t)(i - 9) * VOCAB * DMODEL;
      vmix_k<<<EW, 256, 0, stream>>>(vb, ve, inputs, blk[i].attn_lambdas);
      flash_attn<<<NHEAD * (TSEQ / 16) / 8, 256, 0, stream>>>(qb, kb, vb, yb, docs, winp);
      gemm_xwT<<<g768, 256, 0, stream>>>(yb, blk[i].wo, tmp, TSEQ, DMODEL, DMODEL);
      add_k<<<EW, 256, 0, stream>>>(x, tmp, (int)TD);
    }
    rmsnorm_k<<<TSEQ, 256, 0, stream>>>(x, xn, DMODEL);
    gemm_xwT<<<g3072, 256, 0, stream>>>(xn, blk[i].w_fc, hb, TSEQ, 4 * DMODEL, DMODEL);
    relu2_k<<<EW * 4, 256, 0, stream>>>(hb, (int)(TD * 4));
    gemm_xwT<<<g768, 256, 0, stream>>>(hb, blk[i].w_fcproj, tmp, TSEQ, DMODEL, 4 * DMODEL);
    add_k<<<EW, 256, 0, stream>>>(x, tmp, (int)TD);
    if (i < 6)
      hipMemcpyAsync(skips[i], x, TD * 4, hipMemcpyDeviceToDevice, stream);
  }

  rmsnorm_k<<<TSEQ, 256, 0, stream>>>(x, xn, DMODEL);
  gemm_xwT<<<gvp, 256, 0, stream>>>(xn, lmhead, logits, TSEQ, VPAD, DMODEL);
  loss_k<<<TSEQ, 256, 0, stream>>>(logits, targets, (float*)d_out);
}
